// ConceptNet_59064390254999
// MI455X (gfx1250) — matrix operand order repaired and checked by reading, not compiled, not run on hardware
//
#include <hip/hip_runtime.h>

// ---------------------------------------------------------------------------
// MI455X (gfx1250) ConceptNet pipeline, bf16 WMMA with f32 accumulate.
//  - A tiles row-major in LDS, B tiles K-innermost (transposed) in LDS so
//    every WMMA fragment load is 2x ds_load_b128 (per ISA 7.12.2 layouts).
//  - LDS K-stride padded to 40 bf16 (80B) to avoid 64B bank stride.
//  - cx_dot GEMM fused with per-tile top-10; deterministic serial finalize.
// ---------------------------------------------------------------------------

typedef __bf16 bf16_t;
typedef __attribute__((ext_vector_type(16))) __bf16 v16bf;
typedef __attribute__((ext_vector_type(8)))  __bf16 v8bf;
typedef __attribute__((ext_vector_type(8)))  float  v8f;

#define THRES_V 0.1f
#define KNN 10
#define BKP 40  // padded LDS K-stride (bf16 elems), 80B, 16B-aligned

// ---- WMMA fragment helpers (wave32) ----------------------------------------

__device__ __forceinline__ v16bf frag_a(const bf16_t* As, int mbase) {
  // A 16x32 bf16: lane%16 = M row; per lane two contiguous K-runs:
  // k = half*8 .. +7  and  k = 16+half*8 .. +7  -> two 16B LDS loads
  const int lane = threadIdx.x & 31;
  const int half = lane >> 4;
  const bf16_t* row = As + (mbase + (lane & 15)) * BKP + (half << 3);
  union { v16bf v; v8bf h[2]; } u;
  u.h[0] = *(const v8bf*)(row);
  u.h[1] = *(const v8bf*)(row + 16);
  return u.v;
}

__device__ __forceinline__ v16bf frag_bt(const bf16_t* Bs, int nbase) {
  // B 32x16 bf16 from K-innermost LDS tile: lane%16 = N col;
  // k = half*8 .. +7  and  k = 16+half*8 .. +7  -> two 16B LDS loads
  const int lane = threadIdx.x & 31;
  const int half = lane >> 4;
  const bf16_t* row = Bs + (nbase + (lane & 15)) * BKP + (half << 3);
  union { v16bf v; v8bf h[2]; } u;
  u.h[0] = *(const v8bf*)(row);
  u.h[1] = *(const v8bf*)(row + 16);
  return u.v;
}

__device__ __forceinline__ v8f wmma_bf16(v16bf a, v16bf b, v8f c) {
  return __builtin_amdgcn_wmma_f32_16x16x32_bf16(false, a, false, b,
                                                 (short)0, c, false, false);
}

// ---- Generic tiled GEMM: C[M,N] = A[M,K] * B[K,N] --------------------------
// A: fp32 rows (a_rk==1) or contiguous bf16 rows. B: fp32 strided (b_rk,b_rn).
// EPI 0: f32 store. EPI 1: threshold epilogue. EPI 2: relu -> bf16.

#define GBM 128
#define GBN 64
#define GBK 32

template <int EPI, bool ABF16>
__global__ __launch_bounds__(256) void gemm_wmma(
    const void* __restrict__ Ap, int a_rm,
    const float* __restrict__ Bp, int b_rk, int b_rn,
    float* __restrict__ Cout, bf16_t* __restrict__ CoutB, int c_ld,
    int M, int N, int K,
    const float* __restrict__ rinv, const float* __restrict__ cinv) {
  __shared__ bf16_t As[GBM * BKP];   // row-major [M][Kpad]
  __shared__ bf16_t Bs[GBN * BKP];   // transposed [N][Kpad]
  const int tile_m = blockIdx.y * GBM;
  const int tile_n = blockIdx.x * GBN;
  const int tid  = threadIdx.x;
  const int wave = tid >> 5;
  const int wm = (wave & 3) << 5;   // 4 waves along M
  const int wn = (wave >> 2) << 5;  // 2 waves along N

  // staging coords: A: one 16-elem K-run per thread; B: one 8-elem K-run
  const int ar = tid >> 1, ac = (tid & 1) << 4;
  const int bn = tid & 63, bk = (tid >> 6) << 3;

  v8f acc[2][2] = {};

  for (int k0 = 0; k0 < K; k0 += GBK) {
    {
      bf16_t* dst = As + ar * BKP + ac;
      if (ABF16) {
        const bf16_t* src = (const bf16_t*)Ap + (size_t)(tile_m + ar) * a_rm + (k0 + ac);
#pragma unroll
        for (int j = 0; j < 16; ++j) dst[j] = src[j];
      } else {
        const float* src = (const float*)Ap + (size_t)(tile_m + ar) * a_rm + (k0 + ac);
#pragma unroll
        for (int j = 0; j < 16; ++j) dst[j] = (bf16_t)src[j];
      }
    }
    {
      bf16_t* dst = Bs + bn * BKP + bk;
      const float* src = Bp + (size_t)(k0 + bk) * b_rk + (size_t)(tile_n + bn) * b_rn;
#pragma unroll
      for (int j = 0; j < 8; ++j) dst[j] = (bf16_t)src[(size_t)j * b_rk];
    }
    __syncthreads();
    v16bf a0 = frag_a(As, wm);
    v16bf a1 = frag_a(As, wm + 16);
    v16bf b0 = frag_bt(Bs, wn);
    v16bf b1 = frag_bt(Bs, wn + 16);
    acc[0][0] = wmma_bf16(a0, b0, acc[0][0]);
    acc[0][1] = wmma_bf16(a0, b1, acc[0][1]);
    acc[1][0] = wmma_bf16(a1, b0, acc[1][0]);
    acc[1][1] = wmma_bf16(a1, b1, acc[1][1]);
    __syncthreads();
  }

  const int lane = tid & 31, half = lane >> 4, nn = lane & 15;
#pragma unroll
  for (int it = 0; it < 2; ++it)
#pragma unroll
    for (int jt = 0; jt < 2; ++jt)
#pragma unroll
      for (int v = 0; v < 8; ++v) {
        const int m = tile_m + wm + (it << 4) + v + (half << 3);
        const int n = tile_n + wn + (jt << 4) + nn;
        const float val = acc[it][jt][v];
        if (EPI == 0) {
          Cout[(size_t)m * c_ld + n] = val;
        } else if (EPI == 1) {
          const float sn = val * rinv[m] * cinv[n];
          Cout[(size_t)m * c_ld + n] = (sn > THRES_V) ? val : 0.f;
        } else {
          CoutB[(size_t)m * c_ld + n] = (bf16_t)fmaxf(val, 0.f);
        }
      }
}

// ---- cx_dot GEMM fused with per-tile top-10 --------------------------------
// A(m=concept c, k=d) = Wc[d*C + m]; B(k=d, n) = samp[d*N + n].
// Tile 64(M) x 128(N), K step 32. LDS: 5K + 10K + 32K = 47KB.

#define XBM 64
#define XBN 128
#define XBK 32

__global__ __launch_bounds__(256) void cxdot_topk(
    const float* __restrict__ Wc, const float* __restrict__ samp,
    const float* __restrict__ cnormsq, const float* __restrict__ enorm,
    float* __restrict__ candD, float* __restrict__ candV,
    int C, int N, int D) {
  __shared__ bf16_t As[XBM * BKP];   // row-major [M][Kpad]
  __shared__ bf16_t Bs[XBN * BKP];   // transposed [N][Kpad]
  __shared__ float  Ds[XBM * XBN];
  const int tile_m = blockIdx.y * XBM;
  const int tile_n = blockIdx.x * XBN;
  const int tid  = threadIdx.x;
  const int wave = tid >> 5;
  const int wm = (wave & 1) << 5;   // 2 waves along M
  const int wn = (wave >> 1) << 5;  // 4 waves along N

  // staging coords: A: 8-elem K-run per thread; B: 16-elem K-run per thread
  const int am = tid & 63, ak = (tid >> 6) << 3;
  const int bn = tid & 127, bk = (tid >> 7) << 4;

  v8f acc[2][2] = {};

  for (int k0 = 0; k0 < D; k0 += XBK) {
    {
      bf16_t* dst = As + am * BKP + ak;
#pragma unroll
      for (int j = 0; j < 8; ++j)
        dst[j] = (bf16_t)Wc[(size_t)(k0 + ak + j) * C + (tile_m + am)];
    }
    {
      bf16_t* dst = Bs + bn * BKP + bk;
#pragma unroll
      for (int j = 0; j < 16; ++j)
        dst[j] = (bf16_t)samp[(size_t)(k0 + bk + j) * N + (tile_n + bn)];
    }
    __syncthreads();
    v16bf a0 = frag_a(As, wm);
    v16bf a1 = frag_a(As, wm + 16);
    v16bf b0 = frag_bt(Bs, wn);
    v16bf b1 = frag_bt(Bs, wn + 16);
    acc[0][0] = wmma_bf16(a0, b0, acc[0][0]);
    acc[0][1] = wmma_bf16(a0, b1, acc[0][1]);
    acc[1][0] = wmma_bf16(a1, b0, acc[1][0]);
    acc[1][1] = wmma_bf16(a1, b1, acc[1][1]);
    __syncthreads();
  }

  const int lane = tid & 31, half = lane >> 4, nn = lane & 15;
#pragma unroll
  for (int it = 0; it < 2; ++it)
#pragma unroll
    for (int jt = 0; jt < 2; ++jt)
#pragma unroll
      for (int v = 0; v < 8; ++v)
        Ds[(wm + (it << 4) + v + (half << 3)) * XBN + (wn + (jt << 4) + nn)] =
            acc[it][jt][v];
  __syncthreads();

  if (tid < XBM) {
    const int row = tile_m + tid;
    const float cn = cnormsq[row];
    float bd[KNN], bv[KNN];
#pragma unroll
    for (int i = 0; i < KNN; ++i) { bd[i] = 3.4e38f; bv[i] = 0.f; }
    for (int j = 0; j < XBN; ++j) {
      const float dot  = Ds[tid * XBN + j];
      const float dist = cn + enorm[tile_n + j] - 2.f * dot;
      if (dist < bd[KNN - 1]) {
        int p = KNN - 1;
        while (p > 0 && bd[p - 1] > dist) {
          bd[p] = bd[p - 1]; bv[p] = bv[p - 1]; --p;
        }
        bd[p] = dist; bv[p] = dot;
      }
    }
    const size_t base = ((size_t)row * gridDim.x + blockIdx.x) * KNN;
#pragma unroll
    for (int i = 0; i < KNN; ++i) { candD[base + i] = bd[i]; candV[base + i] = bv[i]; }
  }
}

// ---- small helper kernels --------------------------------------------------

__global__ void col_norms_kernel(const float* __restrict__ W, float* cnormsq,
                                 float* cinv, int D, int C) {
  const int c = blockIdx.x * blockDim.x + threadIdx.x;
  if (c >= C) return;
  float s = 0.f;
  for (int d = 0; d < D; ++d) { const float w = W[(size_t)d * C + c]; s += w * w; }
  cnormsq[c] = s;
  cinv[c] = 1.f / fmaxf(sqrtf(s), 1e-12f);
}

__global__ void row_inv_norm_kernel(const float* __restrict__ X, float* rinv,
                                    int Bn, int D) {
  const int wave = threadIdx.x >> 5, lane = threadIdx.x & 31;
  const int row = blockIdx.x * 8 + wave;
  if (row >= Bn) return;
  const float* x = X + (size_t)row * D;
  float s = 0.f;
  for (int i = lane; i < D; i += 32) { const float v = x[i]; s += v * v; }
  for (int o = 16; o > 0; o >>= 1) s += __shfl_xor(s, o, 32);
  if (lane == 0) rinv[row] = 1.f / fmaxf(sqrtf(s), 1e-12f);
}

__global__ void enorm_kernel(const float* __restrict__ S, float* enorm,
                             int D, int N) {
  const int n = blockIdx.x * blockDim.x + threadIdx.x;
  if (n >= N) return;
  float s = 0.f;
  for (int d = 0; d < D; ++d) { const float v = S[(size_t)d * N + n]; s += v * v; }
  enorm[n] = s;
}

__global__ void l2_kernel(const float* __restrict__ W, float* l2rs, float* l2ss,
                          int D, int C) {
  __shared__ float rs_s[256], ss_s[256];
  const int d = blockIdx.x;
  float rs = 0.f, ss = 0.f;
  for (int c = threadIdx.x; c < C; c += blockDim.x) {
    const float w = W[(size_t)d * C + c];
    rs += w; ss += w * w;
  }
  rs_s[threadIdx.x] = rs; ss_s[threadIdx.x] = ss;
  __syncthreads();
  for (int o = 128; o > 0; o >>= 1) {
    if ((int)threadIdx.x < o) {
      rs_s[threadIdx.x] += rs_s[threadIdx.x + o];
      ss_s[threadIdx.x] += ss_s[threadIdx.x + o];
    }
    __syncthreads();
  }
  if (threadIdx.x == 0) { l2rs[d] = rs_s[0] * rs_s[0]; l2ss[d] = ss_s[0]; }
}

__global__ void row_prob_kernel(const float* __restrict__ T,
                                float* __restrict__ prob,
                                bf16_t* __restrict__ probB, int Bn, int C) {
  const int wave = threadIdx.x >> 5, lane = threadIdx.x & 31;
  const int row = blockIdx.x * 8 + wave;
  if (row >= Bn) return;
  const float* t = T + (size_t)row * C;
  float vals[16];  // C == 512 -> 16 per lane
  float s = 0.f;
#pragma unroll
  for (int i = 0; i < 16; ++i) { vals[i] = t[lane + (i << 5)]; s += vals[i]; }
  for (int o = 16; o > 0; o >>= 1) s += __shfl_xor(s, o, 32);
  const float inv = 1.f / (s + 0.001f);
#pragma unroll
  for (int i = 0; i < 16; ++i) {
    const float p = vals[i] * inv;
    prob[(size_t)row * C + lane + (i << 5)]  = p;
    probB[(size_t)row * C + lane + (i << 5)] = (bf16_t)p;
  }
}

__global__ void knn_merge(const float* __restrict__ candD,
                          const float* __restrict__ candV,
                          float* __restrict__ rowdot, int C, int NT) {
  const int row = blockIdx.x * blockDim.x + threadIdx.x;
  if (row >= C) return;
  float bd[KNN], bv[KNN];
#pragma unroll
  for (int i = 0; i < KNN; ++i) { bd[i] = 3.4e38f; bv[i] = 0.f; }
  const float* dp = candD + (size_t)row * NT * KNN;
  const float* vp = candV + (size_t)row * NT * KNN;
  for (int i = 0; i < NT * KNN; ++i) {
    const float dist = dp[i];
    if (dist < bd[KNN - 1]) {
      const float dot = vp[i];
      int p = KNN - 1;
      while (p > 0 && bd[p - 1] > dist) { bd[p] = bd[p - 1]; bv[p] = bv[p - 1]; --p; }
      bd[p] = dist; bv[p] = dot;
    }
  }
  float s = 0.f;
#pragma unroll
  for (int i = 0; i < KNN; ++i) s += bv[i];
  rowdot[row] = s;
}

__global__ void finalize_kernel(const float* __restrict__ rowdot,
                                const float* __restrict__ l2rs,
                                const float* __restrict__ l2ss,
                                float* outL1, float* outL2, int C, int D, int k) {
  if (threadIdx.x != 0 || blockIdx.x != 0) return;
  float s1 = 0.f;
  for (int c = 0; c < C; ++c) s1 += rowdot[c];
  outL1[0] = s1 / (float)(C * k);
  float a = 0.f, b = 0.f;
  for (int d = 0; d < D; ++d) { a += l2rs[d]; b += l2ss[d]; }
  outL2[0] = (a - b) / (float)(C * C);
}

// ---------------------------------------------------------------------------

extern "C" void kernel_launch(void* const* d_in, const int* in_sizes, int n_in,
                              void* d_out, int out_size, void* d_ws,
                              size_t ws_size, hipStream_t stream) {
  (void)in_sizes; (void)n_in; (void)out_size; (void)ws_size;
  const float* emb  = (const float*)d_in[0];  // [8192][768]
  const float* samp = (const float*)d_in[1];  // [768][65536]
  const float* Wc   = (const float*)d_in[2];  // [768][512]
  const float* R1   = (const float*)d_in[3];  // [512][256]
  const float* R2   = (const float*)d_in[4];  // [256][768]

  const int B = 8192, D = 768, C = 512, H = 256, N = 65536;
  const int NT = N / XBN;  // 512 candidate tiles per row

  char* w = (char*)d_ws;
  auto carve = [&](size_t bytes) {
    char* p = w;
    w += (bytes + 255) & ~(size_t)255;
    return p;
  };
  float*  thresS  = (float*)carve((size_t)B * C * 4);
  bf16_t* probB   = (bf16_t*)carve((size_t)B * C * 2);
  bf16_t* rec1B   = (bf16_t*)carve((size_t)B * H * 2);
  float*  cnormsq = (float*)carve((size_t)C * 4);
  float*  cinv    = (float*)carve((size_t)C * 4);
  float*  rinv    = (float*)carve((size_t)B * 4);
  float*  enorm   = (float*)carve((size_t)N * 4);
  float*  candD   = (float*)carve((size_t)C * NT * KNN * 4);
  float*  candV   = (float*)carve((size_t)C * NT * KNN * 4);
  float*  rowdot  = (float*)carve((size_t)C * 4);
  float*  l2rs    = (float*)carve((size_t)D * 4);
  float*  l2ss    = (float*)carve((size_t)D * 4);

  float* out      = (float*)d_out;
  float* out_rec2 = out;                                  // [B][D]
  float* out_prob = out + (size_t)B * D;                  // [B][C]
  float* out_L1   = out + (size_t)B * D + (size_t)B * C;  // scalar
  float* out_L2   = out_L1 + 1;                           // scalar

  // norms + L2 partials (independent)
  col_norms_kernel<<<C / 256, 256, 0, stream>>>(Wc, cnormsq, cinv, D, C);
  row_inv_norm_kernel<<<B / 8, 256, 0, stream>>>(emb, rinv, B, D);
  enorm_kernel<<<N / 256, 256, 0, stream>>>(samp, enorm, D, N);
  l2_kernel<<<D, 256, 0, stream>>>(Wc, l2rs, l2ss, D, C);

  // S = emb @ concept with fused normalize+threshold epilogue
  gemm_wmma<1, false><<<dim3(C / GBN, B / GBM), 256, 0, stream>>>(
      emb, D, Wc, C, 1, thresS, nullptr, C, B, C, D, rinv, cinv);
  // row sums -> prob (f32 to out, bf16 to ws)
  row_prob_kernel<<<B / 8, 256, 0, stream>>>(thresS, out_prob, probB, B, C);
  // rec1 = relu(prob @ R1), bf16 out
  gemm_wmma<2, true><<<dim3(H / GBN, B / GBM), 256, 0, stream>>>(
      probB, C, R1, H, 1, nullptr, rec1B, H, B, H, C, nullptr, nullptr);
  // rec2 = rec1 @ R2, f32 to out
  gemm_wmma<0, true><<<dim3(D / GBN, B / GBM), 256, 0, stream>>>(
      rec1B, H, R2, D, 1, out_rec2, nullptr, D, B, D, H, nullptr, nullptr);

  // cx_dot + fused per-tile top-10, then merge + finalize (deterministic)
  cxdot_topk<<<dim3(NT, C / XBM), 256, 0, stream>>>(Wc, samp, cnormsq, enorm,
                                                    candD, candV, C, N, D);
  knn_merge<<<(C + 255) / 256, 256, 0, stream>>>(candD, candV, rowdot, C, NT);
  finalize_kernel<<<1, 1, 0, stream>>>(rowdot, l2rs, l2ss, out_L1, out_L2,
                                       C, D, KNN);
}
